// SSMoE_MVMS_Retrieval_8495445311576
// MI455X (gfx1250) — matrix operand order repaired and checked by reading, not compiled, not run on hardware
//
#include <hip/hip_runtime.h>
#include <hip/hip_bf16.h>
#include <math.h>

// ---------------- model dims ----------------
#define D_MODEL 768
#define T_TOT   1024          // B*L = 4*256
#define SEQ     256
#define NHEAD   12
#define HDIM    64
#define NLAYER  2
#define NEXP    8
#define I_EXP   1536
#define I_SH    3072
#define CLH     256

typedef _Float16 v16h __attribute__((ext_vector_type(16)));
typedef _Float16 v8h  __attribute__((ext_vector_type(8)));
typedef float    v8f  __attribute__((ext_vector_type(8)));

constexpr int GEMM_LDT = 40;   // half-stride per LDS tile row: 80B, 16B aligned

// ---- CDNA5 async copy: global f16 -> LDS, 16B per lane, tracked by ASYNCcnt ----
__device__ __forceinline__ void async_copy_b128(_Float16* lds, const _Float16* g) {
  // Flat address of a __shared__ object: low 32 bits are the LDS byte offset
  // (ISA 10.2 aperture mapping).
  unsigned l = (unsigned)(uintptr_t)lds;
  asm volatile("global_load_async_to_lds_b128 %0, %1, off"
               :: "v"(l), "v"(g)
               : "memory");
}
__device__ __forceinline__ void wait_asynccnt0() {
  asm volatile("s_wait_asynccnt 0x0" ::: "memory");
}

// ================= WMMA fragment loaders (CDNA5 16-bit layouts) =================
// A (16x32, MxK): lane holds row M=lane&15; halves: e<8 -> K=(lane&16?8:0)+e,
//                 e>=8 -> K=(lane&16?24:16)+(e-8). Two contiguous 8-half runs.
__device__ __forceinline__ v16h frag_a_from_lds(const _Float16* rowp, int lane) {
  const int lo = (lane & 16) ? 8 : 0;
  v8h a0 = *(const v8h*)(rowp + lo);
  v8h a1 = *(const v8h*)(rowp + lo + 16);
  return __builtin_shufflevector(a0, a1, 0,1,2,3,4,5,6,7,8,9,10,11,12,13,14,15);
}
// B (32x16, KxN): lane holds col N=lane&15; e<8 -> K=(lane&16?8:0)+e,
//                 e>=8 -> K=(lane&16?24:16)+(e-8).
__device__ __forceinline__ v16h frag_b_from_lds(const _Float16* rowp, int lane) {
  const int kb = (lane & 16) ? 8 : 0;
  v8h b0 = *(const v8h*)(rowp + kb);
  v8h b1 = *(const v8h*)(rowp + kb + 16);
  return __builtin_shufflevector(b0, b1, 0,1,2,3,4,5,6,7,8,9,10,11,12,13,14,15);
}

// stage a 64x32 f16 A-tile into LDS via async DMA (2 x b128 per thread)
__device__ __forceinline__ void stage_a_async(const _Float16* __restrict__ A, int K,
                                              int bm, int k0, int tid,
                                              _Float16* sabuf) {
  int c = tid;
  #pragma unroll
  for (int it = 0; it < 2; ++it) {
    const int r = c >> 2, col = (c & 3) * 8;
    async_copy_b128(sabuf + r * GEMM_LDT + col,
                    A + (size_t)(bm + r) * K + k0 + col);
    c += 128;
  }
}

// ================= WMMA GEMM: C[M,N] = A[M,K](f16) x W[N,K]^T(f32) + epilogue ===
// EPI=0: C(f32) = acc + bias[n]            (bias may be null)
// EPI=1: C(f32) += rowscale[m*rs_stride]*acc   (rowscale null -> 1.0)
// EPI=2: C(f16) = acc
// M,N multiples of 64; K multiple of 32. Block = 128 thr (4 waves), 64x64 tile.
// Double-buffered LDS; A staged with GLOBAL_LOAD_ASYNC_TO_LDS_B128, W converted
// f32->f16 inline through VGPRs (roofline-optimal: W is read exactly once).
template <int EPI>
__global__ __launch_bounds__(128)
void gemm_f16_wmma(const _Float16* __restrict__ A, const float* __restrict__ W,
                   const float* __restrict__ bias, const float* __restrict__ rowscale,
                   int rs_stride, float* __restrict__ C, int M, int N, int K)
{
  constexpr int LDT = GEMM_LDT;
  __shared__ alignas(16) _Float16 sa[2][64 * LDT];
  __shared__ alignas(16) _Float16 sb[2][64 * LDT];

  const int tid  = threadIdx.x;
  const int lane = tid & 31;
  const int wave = tid >> 5;
  const int wm   = (wave >> 1) * 32;
  const int wn   = (wave & 1)  * 32;
  const int bm   = blockIdx.y * 64;
  const int bn   = blockIdx.x * 64;
  const int sr   = tid >> 1;                    // staging row 0..63 (W tile)
  const int sc   = (tid & 1) * 16;              // staging col 0 / 16

  v8f zero;
  #pragma unroll
  for (int i = 0; i < 8; ++i) zero[i] = 0.0f;
  v8f acc00 = zero, acc01 = zero, acc10 = zero, acc11 = zero;

  const float* gw = W + (size_t)(bn + sr) * K + sc;

  auto store_w = [&](int b, const float4& w0, const float4& w1,
                     const float4& w2, const float4& w3) {
    v8h h0, h1;
    h0[0]=(_Float16)w0.x; h0[1]=(_Float16)w0.y; h0[2]=(_Float16)w0.z; h0[3]=(_Float16)w0.w;
    h0[4]=(_Float16)w1.x; h0[5]=(_Float16)w1.y; h0[6]=(_Float16)w1.z; h0[7]=(_Float16)w1.w;
    h1[0]=(_Float16)w2.x; h1[1]=(_Float16)w2.y; h1[2]=(_Float16)w2.z; h1[3]=(_Float16)w2.w;
    h1[4]=(_Float16)w3.x; h1[5]=(_Float16)w3.y; h1[6]=(_Float16)w3.z; h1[7]=(_Float16)w3.w;
    _Float16* lb = sb[b] + sr * LDT + sc;
    *(v8h*)(lb)     = h0;
    *(v8h*)(lb + 8) = h1;
  };

  const int nsteps = K >> 5;

  // ---- prologue: stage tile 0 into buffer 0 ----
  stage_a_async(A, K, bm, 0, tid, sa[0]);
  {
    const float4* pw = (const float4*)gw;
    float4 w0 = pw[0], w1 = pw[1], w2 = pw[2], w3 = pw[3];
    store_w(0, w0, w1, w2, w3);
  }
  wait_asynccnt0();
  __syncthreads();

  for (int s = 0; s < nsteps; ++s) {
    const int  buf = s & 1;
    const bool nxt = (s + 1 < nsteps);
    float4 w0, w1, w2, w3;
    if (nxt) {
      // overlap next tile's DMA + global loads with this tile's WMMAs
      stage_a_async(A, K, bm, (s + 1) * 32, tid, sa[buf ^ 1]);
      const float4* pw = (const float4*)(gw + (s + 1) * 32);
      w0 = pw[0]; w1 = pw[1]; w2 = pw[2]; w3 = pw[3];
      if (s + 2 < nsteps) __builtin_prefetch(gw + (s + 2) * 32, 0, 1);
    }

    const int r15 = lane & 15;
    const _Float16* sab = sa[buf];
    const _Float16* sbb = sb[buf];
    v16h af0 = frag_a_from_lds(sab + (wm      + r15) * LDT, lane);
    v16h af1 = frag_a_from_lds(sab + (wm + 16 + r15) * LDT, lane);
    v16h bf0 = frag_b_from_lds(sbb + (wn      + r15) * LDT, lane);
    v16h bf1 = frag_b_from_lds(sbb + (wn + 16 + r15) * LDT, lane);

    acc00 = __builtin_amdgcn_wmma_f32_16x16x32_f16(false, af0, false, bf0, (short)0, acc00, false, false);
    acc01 = __builtin_amdgcn_wmma_f32_16x16x32_f16(false, af0, false, bf1, (short)0, acc01, false, false);
    acc10 = __builtin_amdgcn_wmma_f32_16x16x32_f16(false, af1, false, bf0, (short)0, acc10, false, false);
    acc11 = __builtin_amdgcn_wmma_f32_16x16x32_f16(false, af1, false, bf1, (short)0, acc11, false, false);

    if (nxt) {
      store_w(buf ^ 1, w0, w1, w2, w3);
      wait_asynccnt0();
      __syncthreads();
    }
  }

  // C/D layout: M = r + 8*(lane>=16), N = lane&15 (per 16x16 tile)
  const int nn0 = bn + wn + (lane & 15);
  const int nn1 = nn0 + 16;
  const int mb  = bm + wm + ((lane >> 4) << 3);
  float bv0 = 0.f, bv1 = 0.f;
  if (EPI == 0 && bias) { bv0 = bias[nn0]; bv1 = bias[nn1]; }
  _Float16* C16 = (_Float16*)C;
  #pragma unroll
  for (int r = 0; r < 8; ++r) {
    const int m0 = mb + r;
    const int m1 = mb + 16 + r;
    if (EPI == 0) {
      C[(size_t)m0 * N + nn0] = acc00[r] + bv0;
      C[(size_t)m0 * N + nn1] = acc01[r] + bv1;
      C[(size_t)m1 * N + nn0] = acc10[r] + bv0;
      C[(size_t)m1 * N + nn1] = acc11[r] + bv1;
    } else if (EPI == 1) {
      const float s0 = rowscale ? rowscale[(size_t)m0 * rs_stride] : 1.0f;
      const float s1 = rowscale ? rowscale[(size_t)m1 * rs_stride] : 1.0f;
      C[(size_t)m0 * N + nn0] += s0 * acc00[r];
      C[(size_t)m0 * N + nn1] += s0 * acc01[r];
      C[(size_t)m1 * N + nn0] += s1 * acc10[r];
      C[(size_t)m1 * N + nn1] += s1 * acc11[r];
    } else {
      C16[(size_t)m0 * N + nn0] = (_Float16)acc00[r];
      C16[(size_t)m0 * N + nn1] = (_Float16)acc01[r];
      C16[(size_t)m1 * N + nn0] = (_Float16)acc10[r];
      C16[(size_t)m1 * N + nn1] = (_Float16)acc11[r];
    }
  }
}

// ================= embedding + LN (x stays fp32; LN output f16 for WMMA A) =====
__global__ __launch_bounds__(256)
void embed_ln_kernel(const float* __restrict__ graph, const float* __restrict__ smiles,
                     const float* __restrict__ pos_emb, const float* __restrict__ mod_emb,
                     const float* __restrict__ w, const float* __restrict__ b,
                     float* __restrict__ out)
{
  const int tok = blockIdx.x;            // 0..1023
  const int bb  = tok >> 8;
  const int p   = tok & 255;
  const float* src = (p < 128) ? (graph  + ((size_t)bb * 128 + p) * D_MODEL)
                               : (smiles + ((size_t)bb * 128 + (p - 128)) * D_MODEL);
  const float* me = mod_emb + ((p < 128) ? 0 : D_MODEL);
  __shared__ float buf[D_MODEL];
  __shared__ float red[256];
  const int tid = threadIdx.x;

  float s = 0.f;
  for (int d = tid; d < D_MODEL; d += 256) {
    float v = src[d] + pos_emb[(size_t)p * D_MODEL + d] + me[d];
    buf[d] = v; s += v;
  }
  red[tid] = s; __syncthreads();
  for (int o = 128; o > 0; o >>= 1) { if (tid < o) red[tid] += red[tid + o]; __syncthreads(); }
  const float mean = red[0] / (float)D_MODEL;
  __syncthreads();
  float vv = 0.f;
  for (int d = tid; d < D_MODEL; d += 256) { float dd = buf[d] - mean; vv += dd * dd; }
  red[tid] = vv; __syncthreads();
  for (int o = 128; o > 0; o >>= 1) { if (tid < o) red[tid] += red[tid + o]; __syncthreads(); }
  const float inv = rsqrtf(red[0] / (float)D_MODEL + 1e-5f);
  for (int d = tid; d < D_MODEL; d += 256)
    out[(size_t)tok * D_MODEL + d] = (buf[d] - mean) * inv * w[d] + b[d];
}

__global__ __launch_bounds__(256)
void ln_f16_kernel(const float* __restrict__ x, const float* __restrict__ w,
                   const float* __restrict__ b, _Float16* __restrict__ out)
{
  const int tok = blockIdx.x;
  const float* xr = x + (size_t)tok * D_MODEL;
  __shared__ float red[256];
  const int tid = threadIdx.x;
  float s = 0.f;
  for (int d = tid; d < D_MODEL; d += 256) s += xr[d];
  red[tid] = s; __syncthreads();
  for (int o = 128; o > 0; o >>= 1) { if (tid < o) red[tid] += red[tid + o]; __syncthreads(); }
  const float mean = red[0] / (float)D_MODEL;
  __syncthreads();
  float vv = 0.f;
  for (int d = tid; d < D_MODEL; d += 256) { float dd = xr[d] - mean; vv += dd * dd; }
  red[tid] = vv; __syncthreads();
  for (int o = 128; o > 0; o >>= 1) { if (tid < o) red[tid] += red[tid + o]; __syncthreads(); }
  const float inv = rsqrtf(red[0] / (float)D_MODEL + 1e-5f);
  for (int d = tid; d < D_MODEL; d += 256)
    out[(size_t)tok * D_MODEL + d] = (_Float16)((xr[d] - mean) * inv * w[d] + b[d]);
}

// ================= attention (one block per (b,h), online softmax) ==============
__global__ __launch_bounds__(256)
void attention_kernel(const float* __restrict__ qkv, const int* __restrict__ am,
                      _Float16* __restrict__ out)
{
  const int bh = blockIdx.x;
  const int b  = bh / NHEAD;
  const int h  = bh % NHEAD;
  __shared__ _Float16 ks[SEQ * HDIM];
  __shared__ _Float16 vs[SEQ * HDIM];
  __shared__ float    extm[SEQ];
  const int tid = threadIdx.x;

  for (int i = tid; i < SEQ * HDIM; i += 256) {
    const int t = i >> 6, d = i & 63;
    const size_t row = ((size_t)b * SEQ + t) * (3 * D_MODEL);
    ks[i] = (_Float16)qkv[row +     D_MODEL + h * HDIM + d];
    vs[i] = (_Float16)qkv[row + 2 * D_MODEL + h * HDIM + d];
  }
  extm[tid] = (1.0f - (float)am[b * SEQ + tid]) * -10000.0f;
  __syncthreads();

  const int i = tid;                         // query row
  const size_t qrow = ((size_t)b * SEQ + i) * (3 * D_MODEL) + h * HDIM;
  float q[HDIM];
  #pragma unroll
  for (int d = 0; d < HDIM; ++d) q[d] = qkv[qrow + d];

  float mmax = -1e30f, lsum = 0.f;
  float o[HDIM];
  #pragma unroll
  for (int d = 0; d < HDIM; ++d) o[d] = 0.f;

  for (int j = 0; j < SEQ; ++j) {
    float s = 0.f;
    #pragma unroll
    for (int d = 0; d < HDIM; ++d) s += q[d] * (float)ks[j * HDIM + d];
    s = s * 0.125f + extm[j];                // 1/sqrt(64)
    const float nm = fmaxf(mmax, s);
    const float sc = __expf(mmax - nm);
    const float p  = __expf(s - nm);
    lsum = lsum * sc + p;
    #pragma unroll
    for (int d = 0; d < HDIM; ++d) o[d] = o[d] * sc + p * (float)vs[j * HDIM + d];
    mmax = nm;
  }
  const float inv = 1.0f / lsum;
  #pragma unroll
  for (int d = 0; d < HDIM; ++d)
    out[((size_t)b * SEQ + i) * D_MODEL + h * HDIM + d] = (_Float16)(o[d] * inv);
}

// ================= small elementwise / routing kernels ==========================
__global__ void add_kernel(float* __restrict__ x, const float* __restrict__ y, int n) {
  int i = blockIdx.x * blockDim.x + threadIdx.x;
  if (i < n) x[i] += y[i];
}
__global__ void fill0_kernel(float* __restrict__ p, int n) {
  int i = blockIdx.x * blockDim.x + threadIdx.x;
  if (i < n) p[i] = 0.f;
}
__global__ void silu_mul_f16_kernel(const _Float16* __restrict__ g,
                                    const _Float16* __restrict__ u,
                                    _Float16* __restrict__ o, int n) {
  int i = blockIdx.x * blockDim.x + threadIdx.x;
  if (i < n) {
    float x = (float)g[i];
    o[i] = (_Float16)((x / (1.0f + __expf(-x))) * (float)u[i]);
  }
}
__global__ void gate_logits_kernel(const _Float16* __restrict__ h,
                                   const float* __restrict__ gw,
                                   float* __restrict__ logits) {
  int idx = blockIdx.x * blockDim.x + threadIdx.x;    // m*8+e
  if (idx >= T_TOT * NEXP) return;
  const int m = idx >> 3, e = idx & 7;
  const _Float16* hr = h + (size_t)m * D_MODEL;
  const float*    wr = gw + (size_t)e * D_MODEL;
  float s = 0.f;
  for (int d = 0; d < D_MODEL; ++d) s += (float)hr[d] * wr[d];
  logits[idx] = s;
}
__global__ void topk_kernel(const float* __restrict__ logits, float* __restrict__ comb) {
  int m = blockIdx.x * blockDim.x + threadIdx.x;
  if (m >= T_TOT) return;
  float l[NEXP]; float mx = -1e30f;
  for (int e = 0; e < NEXP; ++e) { l[e] = logits[m * NEXP + e]; mx = fmaxf(mx, l[e]); }
  float s = 0.f;
  for (int e = 0; e < NEXP; ++e) { l[e] = __expf(l[e] - mx); s += l[e]; }
  for (int e = 0; e < NEXP; ++e) l[e] /= s;
  int i1 = 0;
  for (int e = 1; e < NEXP; ++e) if (l[e] > l[i1]) i1 = e;
  int i2 = (i1 == 0) ? 1 : 0;
  for (int e = 0; e < NEXP; ++e) if (e != i1 && l[e] > l[i2]) i2 = e;
  const float tot = l[i1] + l[i2] + 1e-20f;
  for (int e = 0; e < NEXP; ++e) comb[m * NEXP + e] = 0.f;
  comb[m * NEXP + i1] = l[i1] / tot;
  comb[m * NEXP + i2] = l[i2] / tot;
}

// ================= pooling + head ===============================================
__global__ __launch_bounds__(256)
void pool_kernel(const float* __restrict__ x, const int* __restrict__ am,
                 const float* __restrict__ pq, float* __restrict__ pooled)
{
  const int b = blockIdx.x;
  __shared__ float pw[SEQ];
  __shared__ float red[256];
  const int t = threadIdx.x;
  const float* xr = x + ((size_t)b * SEQ + t) * D_MODEL;
  float s = 0.f;
  for (int d = 0; d < D_MODEL; ++d) s += xr[d] * pq[d];
  s *= (1.0f / sqrtf((float)D_MODEL));
  if (am[b * SEQ + t] == 0) s = -1e9f;
  red[t] = s; __syncthreads();
  for (int o = 128; o > 0; o >>= 1) { if (t < o) red[t] = fmaxf(red[t], red[t + o]); __syncthreads(); }
  const float mx = red[0]; __syncthreads();
  const float p = __expf(s - mx);
  red[t] = p; __syncthreads();
  for (int o = 128; o > 0; o >>= 1) { if (t < o) red[t] += red[t + o]; __syncthreads(); }
  pw[t] = p / red[0];
  __syncthreads();
  for (int d = t; d < D_MODEL; d += 256) {
    float acc = 0.f;
    for (int tt = 0; tt < SEQ; ++tt) acc += pw[tt] * x[((size_t)b * SEQ + tt) * D_MODEL + d];
    pooled[b * D_MODEL + d] = acc;
  }
}
__global__ __launch_bounds__(256)
void head_kernel(const float* __restrict__ pooled, const float* __restrict__ mw,
                 const float* __restrict__ mb, float* __restrict__ out)
{
  const int b = blockIdx.x;
  const int c = threadIdx.x;                 // 0..255
  const float* pr = pooled + (size_t)b * D_MODEL;
  const float* wr = mw + (size_t)c * D_MODEL;
  float s = 0.f;
  for (int d = 0; d < D_MODEL; ++d) s += pr[d] * wr[d];
  out[b * CLH + c] = s + mb[c];
}

// ================= host orchestration ===========================================
extern "C" void kernel_launch(void* const* d_in, const int* in_sizes, int n_in,
                              void* d_out, int out_size, void* d_ws, size_t ws_size,
                              hipStream_t stream)
{
  const float* graph   = (const float*)d_in[0];
  const float* smiles  = (const float*)d_in[1];
  const int*   attmask = (const int*)  d_in[2];
  const float* pos_emb = (const float*)d_in[3];
  const float* mod_emb = (const float*)d_in[4];
  const float* eln_w   = (const float*)d_in[5];
  const float* eln_b   = (const float*)d_in[6];
  const float* ln1_w   = (const float*)d_in[7];
  const float* ln1_b   = (const float*)d_in[8];
  const float* qkv_w   = (const float*)d_in[9];
  const float* qkv_b   = (const float*)d_in[10];
  const float* proj_w  = (const float*)d_in[11];
  const float* proj_b  = (const float*)d_in[12];
  const float* ln2_w   = (const float*)d_in[13];
  const float* ln2_b   = (const float*)d_in[14];
  const float* gate_w  = (const float*)d_in[15];
  const float* exp_g_w = (const float*)d_in[16];
  const float* exp_u_w = (const float*)d_in[17];
  const float* exp_d_w = (const float*)d_in[18];
  const float* sh_g_w  = (const float*)d_in[19];
  const float* sh_u_w  = (const float*)d_in[20];
  const float* sh_d_w  = (const float*)d_in[21];
  const float* pool_q  = (const float*)d_in[22];
  const float* mol_w   = (const float*)d_in[23];
  const float* mol_b   = (const float*)d_in[24];
  float* out = (float*)d_out;

  // workspace carve-up (slots sized in fp32 elements; f16 views reuse slots)
  float* ws = (float*)d_ws;
  const size_t TD  = (size_t)T_TOT * D_MODEL;        // 786432
  float*     x      = ws;                                  // [1024,768] f32
  _Float16*  h16    = (_Float16*)(x + TD);                 // [1024,768] f16
  float*     qkv    = (float*)(x + 2 * TD);                // [1024,2304] f32
  _Float16*  attn16 = (_Float16*)(qkv + (size_t)T_TOT * 3 * D_MODEL); // f16
  float*     tmp    = (float*)(attn16) + TD;               // [1024,768] f32 (slot TD after attn slot)
  _Float16*  g1     = (_Float16*)(tmp + TD);               // [1024,3072] f16
  _Float16*  g2     = (_Float16*)((float*)g1 + (size_t)T_TOT * I_SH);
  _Float16*  gg     = (_Float16*)((float*)g2 + (size_t)T_TOT * I_SH);
  float*     moe    = (float*)gg + (size_t)T_TOT * I_SH;   // [1024,768] f32
  float*     logits = moe + TD;                            // [1024,8]
  float*     comb   = logits + (size_t)T_TOT * NEXP;       // [1024,8]
  float*     pooled = comb + (size_t)T_TOT * NEXP;         // [4,768]

  auto gemm0 = [&](const _Float16* A, const float* W, const float* bias, float* C,
                   int M, int N, int K) {
    dim3 g(N / 64, M / 64);
    gemm_f16_wmma<0><<<g, 128, 0, stream>>>(A, W, bias, nullptr, 0, C, M, N, K);
  };
  auto gemm1 = [&](const _Float16* A, const float* W, const float* rs, int rss,
                   float* C, int M, int N, int K) {
    dim3 g(N / 64, M / 64);
    gemm_f16_wmma<1><<<g, 128, 0, stream>>>(A, W, nullptr, rs, rss, C, M, N, K);
  };
  auto gemm2 = [&](const _Float16* A, const float* W, _Float16* C,
                   int M, int N, int K) {
    dim3 g(N / 64, M / 64);
    gemm_f16_wmma<2><<<g, 128, 0, stream>>>(A, W, nullptr, nullptr, 0, (float*)C, M, N, K);
  };

  // ---- embedding ----
  embed_ln_kernel<<<T_TOT, 256, 0, stream>>>(graph, smiles, pos_emb, mod_emb,
                                             eln_w, eln_b, x);

  for (int j = 0; j < NLAYER; ++j) {
    // ---- attention block ----
    ln_f16_kernel<<<T_TOT, 256, 0, stream>>>(x, ln1_w + j * D_MODEL, ln1_b + j * D_MODEL, h16);
    gemm0(h16, qkv_w + (size_t)j * 3 * D_MODEL * D_MODEL, qkv_b + (size_t)j * 3 * D_MODEL,
          qkv, T_TOT, 3 * D_MODEL, D_MODEL);
    attention_kernel<<<4 * NHEAD, 256, 0, stream>>>(qkv, attmask, attn16);
    gemm0(attn16, proj_w + (size_t)j * D_MODEL * D_MODEL, proj_b + (size_t)j * D_MODEL,
          tmp, T_TOT, D_MODEL, D_MODEL);
    add_kernel<<<(int)((TD + 255) / 256), 256, 0, stream>>>(x, tmp, (int)TD);

    // ---- MoE block ----
    ln_f16_kernel<<<T_TOT, 256, 0, stream>>>(x, ln2_w + j * D_MODEL, ln2_b + j * D_MODEL, h16);
    gate_logits_kernel<<<(T_TOT * NEXP + 255) / 256, 256, 0, stream>>>(
        h16, gate_w + (size_t)j * NEXP * D_MODEL, logits);
    topk_kernel<<<(T_TOT + 255) / 256, 256, 0, stream>>>(logits, comb);
    fill0_kernel<<<(int)((TD + 255) / 256), 256, 0, stream>>>(moe, (int)TD);

    for (int e = 0; e < NEXP; ++e) {
      const size_t wofs = ((size_t)j * NEXP + e) * (size_t)I_EXP * D_MODEL;
      gemm2(h16, exp_g_w + wofs, g1, T_TOT, I_EXP, D_MODEL);
      gemm2(h16, exp_u_w + wofs, g2, T_TOT, I_EXP, D_MODEL);
      silu_mul_f16_kernel<<<(T_TOT * I_EXP + 255) / 256, 256, 0, stream>>>(
          g1, g2, gg, T_TOT * I_EXP);
      gemm1(gg, exp_d_w + ((size_t)j * NEXP + e) * (size_t)D_MODEL * I_EXP,
            comb + e, NEXP, moe, T_TOT, D_MODEL, I_EXP);
    }
    // shared experts
    const size_t sofs = (size_t)j * I_SH * D_MODEL;
    gemm2(h16, sh_g_w + sofs, g1, T_TOT, I_SH, D_MODEL);
    gemm2(h16, sh_u_w + sofs, g2, T_TOT, I_SH, D_MODEL);
    silu_mul_f16_kernel<<<(T_TOT * I_SH + 255) / 256, 256, 0, stream>>>(
        g1, g2, gg, T_TOT * I_SH);
    gemm1(gg, sh_d_w + (size_t)j * D_MODEL * I_SH, nullptr, 0, moe,
          T_TOT, D_MODEL, I_SH);

    add_kernel<<<(int)((TD + 255) / 256), 256, 0, stream>>>(x, moe, (int)TD);
  }

  // ---- pooling + head ----
  pool_kernel<<<4, 256, 0, stream>>>(x, attmask, pool_q, pooled);
  head_kernel<<<4, 256, 0, stream>>>(pooled, mol_w, mol_b, out);
}
